// SDFMLP_31903017074991
// MI455X (gfx1250) — compile-verified, matrix operand order checked by reading, not run on hardware
//
#include <hip/hip_runtime.h>

typedef __attribute__((ext_vector_type(16))) _Float16 v16h;
typedef __attribute__((ext_vector_type(8)))  _Float16 v8h;
typedef __attribute__((ext_vector_type(8)))  float    v8f;

constexpr int HID     = 64;
constexpr int WAVES   = 8;    // 256 threads / wave32
constexpr int TILES   = 8;    // 16-point tiles per wave
constexpr int HSTRIDE = 72;   // 64 halfs + 8 pad (rows stay 16B aligned, spreads banks)
constexpr int PTS_PER_BLOCK = WAVES * TILES * 16; // 1024

// max(x,0) as a single v_med3_f32 (clamp to [0,+inf)). Inline asm so InstCombine
// cannot fold it back into maxnum(x,0) + NaN-canonicalize pair.
__device__ __forceinline__ float relu1op(float x) {
    float r;
    asm("v_med3_f32 %0, %1, 0, 0x7f800000" : "=v"(r) : "v"(x));
    return r;
}

// Softplus(beta=100) = (1/100)*log1p(exp(100x)), branch-free & minimal:
//   = max(x,0) + (ln2/100) * log2(1 + exp2(-(100*log2e)*|x|))
// |x| and the negate fold into src modifiers of the mul feeding v_exp_f32;
// 1 + exp2(-t) is in (1,2], so raw v_exp_f32/v_log_f32 need no fix-ups.
__device__ __forceinline__ float softplus100(float x) {
    const float e = __builtin_amdgcn_exp2f(__builtin_fabsf(x) * -144.26950408889634f);
    const float l = __builtin_amdgcn_logf(1.0f + e);
    return fmaf(l, 0.006931471805599453f, relu1op(x));
}

// 16-bit A/B operand lane layout (ISA 7.12.2):
//   lanes 0-15 : halfs 0-7 -> K=0..7,   halfs 8-15 -> K=16..23
//   lanes 16-31: halfs 0-7 -> K=8..15,  halfs 8-15 -> K=24..31
__device__ __forceinline__ int kmap(int j, int hi) {
    return ((j < 8) ? j : (j + 8)) + (hi ? 8 : 0);
}

__global__ __launch_bounds__(256)
void sdf_mlp_wmma_kernel(const float* __restrict__ x,
                         const float* __restrict__ W1, const float* __restrict__ b1,
                         const float* __restrict__ W2, const float* __restrict__ b2,
                         const float* __restrict__ W3, const float* __restrict__ b3,
                         const float* __restrict__ W4, const float* __restrict__ b4,
                         float* __restrict__ out, int n)
{
    // Per-wave activation tiles
    __shared__ alignas(16) _Float16 hbuf[WAVES][16][HSTRIDE];          // 18 KB
    // Weights pre-packed in exact per-lane WMMA B-operand layout:
    // entry [.][ot][kb][lane][j] is lane's v16h for output-tile ot, K-slab kb.
    __shared__ alignas(32) _Float16 BW1[4][32][16];                    //  4 KB
    __shared__ alignas(32) _Float16 BW23[2][4][2][32][16];             // 16 KB
    __shared__ float w4s[HID];
    __shared__ float b4s;

    const int tid  = threadIdx.x;
    const int wave = tid >> 5;
    const int lane = tid & 31;
    const int ln16 = lane & 15;
    const int hi   = lane >> 4;          // 0: low half-wave, 1: high half-wave

    // ---------------- cooperative weight pre-pack into LDS -----------------------
    if (tid < HID) w4s[tid] = W4[tid];
    if (tid == 0)  b4s = b4[0];
    {   // W1 [64][3]: 128 entries (4 ot x 32 lanes), K padded 3 -> 32
        if (tid < 128) {
            const int ot = tid >> 5, l = tid & 31, lln = l & 15, lhi = l >> 4;
            const int o = ot * 16 + lln;
            #pragma unroll
            for (int j = 0; j < 16; ++j) {
                const int k = kmap(j, lhi);
                BW1[ot][l][j] = (!lhi && k < 3) ? (_Float16)W1[o * 3 + k] : (_Float16)0.0f;
            }
        }
        // W2/W3 [64][64]: 512 entries, 2 per thread
        #pragma unroll
        for (int e = tid; e < 512; e += 256) {
            const int layer = e >> 8, rem = e & 255;
            const int ot = rem >> 6, kb = (rem >> 5) & 1, l = rem & 31;
            const int lln = l & 15, lhi = l >> 4;
            const float* W = (layer ? W3 : W2) + (ot * 16 + lln) * HID + kb * 32;
            #pragma unroll
            for (int j = 0; j < 16; ++j)
                BW23[layer][ot][kb][l][j] = (_Float16)W[kmap(j, lhi)];
        }
    }
    __syncthreads();

    // Per-lane biases (lane owns output channel ot*16 + ln16)
    float bb1[4], bb2[4], bb3[4];
    #pragma unroll
    for (int t = 0; t < 4; ++t) {
        const int o = t * 16 + ln16;
        bb1[t] = b1[o]; bb2[t] = b2[o]; bb3[t] = b3[o];
    }

    const long long blockBase = (long long)blockIdx.x * PTS_PER_BLOCK;

    for (int t = 0; t < TILES; ++t) {
        const long long base = blockBase + (long long)wave * (TILES * 16) + t * 16;
        if (base >= n) return;

        // ---------------- layer 1: A = x tile [16 x 3] padded to K=32 ------------
        v16h a = {};
        if (!hi) {                       // high half-wave covers K>=8 -> all zero
            const long long row = base + ln16;
            if (row < n) {
                a[0] = (_Float16)x[row * 3 + 0];
                a[1] = (_Float16)x[row * 3 + 1];
                a[2] = (_Float16)x[row * 3 + 2];
            }
        }
        {   // all 4 WMMAs first (no D->A/B deps => hazard slots filled), zero C
            // lowers to the inline-constant SRC2=0 form: no bias broadcast/copies.
            v8f c[4];
            #pragma unroll
            for (int ot = 0; ot < 4; ++ot) {
                const v16h bw = *(const v16h*)&BW1[ot][lane][0];
                c[ot] = __builtin_amdgcn_wmma_f32_16x16x32_f16(false, a, false, bw,
                                                               (short)0, v8f{}, false, false);
            }
            #pragma unroll
            for (int ot = 0; ot < 4; ++ot) {
                const int o = ot * 16 + ln16;
                #pragma unroll
                for (int r = 0; r < 8; ++r)                  // D: lane=N, vgpr r=M(+8 hi)
                    hbuf[wave][hi * 8 + r][o] = (_Float16)softplus100(c[ot][r] + bb1[ot]);
            }
        }

        // ---------------- layers 2 and 3: 16x64 @ 64x64, K = 2 x 32 --------------
        #pragma unroll
        for (int layer = 0; layer < 2; ++layer) {
            // gather A from this wave's tile: contiguous 8-half runs -> ds_load_b128
            const v8h* row = (const v8h*)&hbuf[wave][ln16][0];
            const v8h s0 = row[hi + 0];   // K =  0..7  (lo) /  8..15 (hi)
            const v8h s1 = row[hi + 2];   // K = 16..23 (lo) / 24..31 (hi)
            const v8h s2 = row[hi + 4];   // K = 32..39 (lo) / 40..47 (hi)
            const v8h s3 = row[hi + 6];   // K = 48..55 (lo) / 56..63 (hi)
            const v16h a0 = __builtin_shufflevector(s0, s1, 0,1,2,3,4,5,6,7,8,9,10,11,12,13,14,15);
            const v16h a1 = __builtin_shufflevector(s2, s3, 0,1,2,3,4,5,6,7,8,9,10,11,12,13,14,15);
            v8f c[4];
            #pragma unroll
            for (int ot = 0; ot < 4; ++ot) {
                const v16h bk0 = *(const v16h*)&BW23[layer][ot][0][lane][0];
                const v16h bk1 = *(const v16h*)&BW23[layer][ot][1][lane][0];
                c[ot] = __builtin_amdgcn_wmma_f32_16x16x32_f16(false, a0, false, bk0,
                                                               (short)0, v8f{}, false, false);
                c[ot] = __builtin_amdgcn_wmma_f32_16x16x32_f16(false, a1, false, bk1,
                                                               (short)0, c[ot], false, false);
            }
            #pragma unroll
            for (int ot = 0; ot < 4; ++ot) {
                const float bb = layer ? bb3[ot] : bb2[ot];
                const int o = ot * 16 + ln16;
                #pragma unroll
                for (int r = 0; r < 8; ++r)
                    hbuf[wave][hi * 8 + r][o] = (_Float16)softplus100(c[ot][r] + bb);
            }
        }

        // ---------------- layer 4: [16x64] x [64x1]; both half-waves work --------
        {
            const v8h* hr = (const v8h*)&hbuf[wave][ln16][0];
            const v8h h0 = hr[hi * 4 + 0];
            const v8h h1 = hr[hi * 4 + 1];
            const v8h h2 = hr[hi * 4 + 2];
            const v8h h3 = hr[hi * 4 + 3];
            float acc = 0.0f;
            #pragma unroll
            for (int j = 0; j < 8; ++j) {
                acc = fmaf((float)h0[j], w4s[hi * 32 + j],      acc);
                acc = fmaf((float)h1[j], w4s[hi * 32 + 8 + j],  acc);
                acc = fmaf((float)h2[j], w4s[hi * 32 + 16 + j], acc);
                acc = fmaf((float)h3[j], w4s[hi * 32 + 24 + j], acc);
            }
            acc += __shfl_xor(acc, 16, 32);   // combine the two K-halves
            if (!hi) {
                const long long row = base + ln16;
                if (row < n) out[row] = acc + b4s;
            }
        }
    }
}

extern "C" void kernel_launch(void* const* d_in, const int* in_sizes, int n_in,
                              void* d_out, int out_size, void* d_ws, size_t ws_size,
                              hipStream_t stream) {
    (void)n_in; (void)out_size; (void)d_ws; (void)ws_size;
    const float* x  = (const float*)d_in[0];
    const float* W1 = (const float*)d_in[1];
    const float* b1 = (const float*)d_in[2];
    const float* W2 = (const float*)d_in[3];
    const float* b2 = (const float*)d_in[4];
    const float* W3 = (const float*)d_in[5];
    const float* b3 = (const float*)d_in[6];
    const float* W4 = (const float*)d_in[7];
    const float* b4 = (const float*)d_in[8];
    float* out = (float*)d_out;

    const int n = in_sizes[0] / 3;                 // x is [N, 3]
    const int grid = (n + PTS_PER_BLOCK - 1) / PTS_PER_BLOCK;
    sdf_mlp_wmma_kernel<<<grid, 256, 0, stream>>>(x, W1, b1, W2, b2, W3, b3,
                                                  W4, b4, out, n);
}
